// Mamba4Rec_25220047962710
// MI455X (gfx1250) — matrix operand order repaired and checked by reading, not compiled, not run on hardware
//
#include <hip/hip_runtime.h>
#include <hip/hip_bf16.h>
#include <math.h>

typedef __attribute__((ext_vector_type(16))) _Float16 v16h;
typedef __attribute__((ext_vector_type(8)))  _Float16 v8h;
typedef __attribute__((ext_vector_type(8)))  float    v8f;

#define BB   32
#define LL   256
#define DD   256
#define DIM  512           // DI
#define NST  16
#define RRK  16            // dt_rank
#define MTOK (BB*LL)       // 8192 tokens

__device__ __forceinline__ float sigmoidf_(float x){ return 1.f/(1.f+__expf(-x)); }
__device__ __forceinline__ float siluf_(float x){ return x*sigmoidf_(x); }
__device__ __forceinline__ float geluf_(float x){ return 0.5f*x*(1.f+erff(x*0.70710678118654752f)); }
__device__ __forceinline__ float softplusf_(float x){ return (x>20.f)?x:log1pf(__expf(x)); }

// Two async global->LDS b128 copies; INST_OFFSET applies to both LDS and
// global address (ISA 08 §4.4), so offset:16 stages the second half of a
// lane's contiguous 32-byte fragment. Tracked by ASYNCcnt.
__device__ __forceinline__ void async_b128x2(unsigned lds_off, unsigned long long gaddr){
  asm volatile("global_load_async_to_lds_b128 %0, %1, off\n\t"
               "global_load_async_to_lds_b128 %0, %1, off offset:16"
               :: "v"(lds_off), "v"(gaddr) : "memory");
}

// ---------------- elementwise / conversion kernels ----------------

__global__ __launch_bounds__(256) void k_f32_to_f16(const float* __restrict__ s,
                                                    _Float16* __restrict__ d, long n){
  long i = (long)blockIdx.x*256 + threadIdx.x;
  if (i < n) d[i] = (_Float16)s[i];
}

__global__ __launch_bounds__(256) void k_fill_f16(_Float16* __restrict__ d, long n){
  long i = (long)blockIdx.x*256 + threadIdx.x;
  if (i < n) d[i] = (_Float16)0.f;
}

// fc2_w [256,16] -> f16 [256,32] zero-padded in K so WMMA K=32 works
__global__ __launch_bounds__(256) void k_pad_fc2(const float* __restrict__ s,
                                                 _Float16* __restrict__ d){
  int i = blockIdx.x*256 + threadIdx.x;           // 256*32 = 8192
  if (i >= 256*32) return;
  int n = i >> 5, k = i & 31;
  d[i] = (k < 16) ? (_Float16)s[n*16 + k] : (_Float16)0.f;
}

// ---------------- embedding gather + LayerNorm ----------------

__global__ __launch_bounds__(256) void k_embed_ln(const int* __restrict__ seq,
    const float* __restrict__ emb, const float* __restrict__ w, const float* __restrict__ b,
    float* __restrict__ h, _Float16* __restrict__ hh){
  int t = blockIdx.x, d = threadIdx.x;
  long idx = (long)seq[t];
  float x = emb[idx*DD + d];
  __shared__ float red[256];
  red[d] = x; __syncthreads();
  for (int s=128;s>0;s>>=1){ if (d<s) red[d]+=red[d+s]; __syncthreads(); }
  float mu = red[0]*(1.f/DD); __syncthreads();
  float xm = x - mu;
  red[d] = xm*xm; __syncthreads();
  for (int s=128;s>0;s>>=1){ if (d<s) red[d]+=red[d+s]; __syncthreads(); }
  float var = red[0]*(1.f/DD);
  float y = xm*rsqrtf(var+1e-12f)*w[d] + b[d];
  h [(long)t*DD+d] = y;
  hh[(long)t*DD+d] = (_Float16)y;
}

// h2 = LN(m + hin)
__global__ __launch_bounds__(256) void k_ln_resid(const float* __restrict__ m,
    const float* __restrict__ hin, const float* __restrict__ w, const float* __restrict__ b,
    float* __restrict__ h2, _Float16* __restrict__ h2h){
  int t = blockIdx.x, d = threadIdx.x;
  float x = m[(long)t*DD+d] + hin[(long)t*DD+d];
  __shared__ float red[256];
  red[d] = x; __syncthreads();
  for (int s=128;s>0;s>>=1){ if (d<s) red[d]+=red[d+s]; __syncthreads(); }
  float mu = red[0]*(1.f/DD); __syncthreads();
  float xm = x - mu;
  red[d] = xm*xm; __syncthreads();
  for (int s=128;s>0;s>>=1){ if (d<s) red[d]+=red[d+s]; __syncthreads(); }
  float var = red[0]*(1.f/DD);
  float y = xm*rsqrtf(var+1e-12f)*w[d] + b[d];
  h2 [(long)t*DD+d] = y;
  h2h[(long)t*DD+d] = (_Float16)y;
}

// ---------------- big WMMA GEMM with async-LDS weight staging ----------------
// C[M,N] = act(A[M,K] @ B[N,K]^T + bias) (+resid). Requires M%64==0, N%128==0, K%32==0.
// Block = 128 thr = 4 waves. Block macro-tile: 64 rows (one 16-row M-tile per wave)
// x 128 cols (8 N-tiles, register-blocked per wave). Per 32-K step the 8KB B-slice
// is staged global->LDS with GLOBAL_LOAD_ASYNC_TO_LDS_B128 (4 copies/thread),
// double-buffered, fenced by s_wait_asynccnt + barrier, consumed via ds_load->wmma.
// Last k-step peeled so the steady-state loop is branch-free.
__global__ __launch_bounds__(128) void k_gemm_f16_big(
    const _Float16* __restrict__ A, const _Float16* __restrict__ Bm,
    float* __restrict__ C, _Float16* __restrict__ Ch,
    const float* __restrict__ bias, const float* __restrict__ resid,
    int N, int K, int ldch, int act){
  __shared__ _Float16 smem[2][256*16];              // 2 x 8KB B slices
  const int tid  = threadIdx.x;
  const int wave = tid >> 5, lane = tid & 31;
  const int r    = lane & 15, kg = lane >> 4;
  const int nbN  = N >> 7;
  const int bm   = blockIdx.x / nbN, bn = blockIdx.x - bm*nbN;
  const int tm   = bm*4 + wave;

  // staging map: thread tid stages fragment slots tid and tid+128
  const int sl   = tid & 31;
  const int scol = bn*128 + (tid>>5)*16 + (sl & 15);
  const _Float16* gB = Bm + (size_t)scol*K + (sl>>4)*16;
  const size_t colstep = (size_t)64*K*2;            // +64 cols, in bytes
  const unsigned lbase = (unsigned)(uintptr_t)&smem[0][0] + (unsigned)tid*32u;

  const _Float16* ap = A + (size_t)(tm*16 + r)*K + kg*16;

  v8f acc0={},acc1={},acc2={},acc3={},acc4={},acc5={},acc6={},acc7={};
  const int nk = K >> 5;

  auto stage = [&](int buf, int ks){
    unsigned lo = lbase + (unsigned)buf*8192u;
    unsigned long long ga = (unsigned long long)(uintptr_t)(gB + (size_t)ks*32);
    async_b128x2(lo,          ga);
    async_b128x2(lo + 4096u,  ga + colstep);
  };

  #define KSTEP(CUR, K0) { \
    v8h a0 = *(const v8h*)(ap + (K0)); \
    v8h a1 = *(const v8h*)(ap + (K0) + 8); \
    v16h av = __builtin_shufflevector(a0, a1, 0,1,2,3,4,5,6,7,8,9,10,11,12,13,14,15); \
    const _Float16* sb = smem[CUR]; \
    BTILE(0, acc0) BTILE(1, acc1) BTILE(2, acc2) BTILE(3, acc3) \
    BTILE(4, acc4) BTILE(5, acc5) BTILE(6, acc6) BTILE(7, acc7) }
  #define BTILE(T, ACC) { \
    const _Float16* q = sb + ((T)*32 + lane)*16; \
    v8h b0 = *(const v8h*)q; \
    v8h b1 = *(const v8h*)(q + 8); \
    v16h bv = __builtin_shufflevector(b0, b1, 0,1,2,3,4,5,6,7,8,9,10,11,12,13,14,15); \
    ACC = __builtin_amdgcn_wmma_f32_16x16x32_f16(false, av, false, bv, \
                                                 (short)0, ACC, false, false); }

  stage(0, 0);                                      // prologue: k-slice 0
  for (int ks = 0; ks < nk - 1; ks++){              // branch-free steady state
    stage((ks + 1) & 1, ks + 1);
    asm volatile("s_wait_asynccnt 0x4" ::: "memory"); // current slice landed
    __syncthreads();                                  // all waves' stages visible
    KSTEP(ks & 1, ks*32)
    __syncthreads();                                  // reads done before re-stage
  }
  asm volatile("s_wait_asynccnt 0x0" ::: "memory");   // final slice
  __syncthreads();
  KSTEP((nk - 1) & 1, (nk - 1)*32)
  #undef BTILE
  #undef KSTEP

  #define EPILOG(T, ACC) { \
    const int col = bn*128 + (T)*16 + r; \
    const float bi = bias ? bias[col] : 0.f; \
    _Pragma("unroll") \
    for (int j = 0; j < 8; j++){ \
      int row = tm*16 + j + 8*kg; \
      float v = ACC[j] + bi; \
      if (act == 1) v = geluf_(v); \
      if (resid)    v += resid[(long)row*N + col]; \
      C[(long)row*N + col] = v; \
      if (Ch) Ch[(long)row*ldch + col] = (_Float16)v; \
    } }
  EPILOG(0, acc0) EPILOG(1, acc1) EPILOG(2, acc2) EPILOG(3, acc3)
  EPILOG(4, acc4) EPILOG(5, acc5) EPILOG(6, acc6) EPILOG(7, acc7)
  #undef EPILOG
}

// ---------------- small WMMA GEMM (N=48 / N=16), direct global loads ----------------

__global__ __launch_bounds__(128) void k_gemm_f16(
    const _Float16* __restrict__ A, const _Float16* __restrict__ Bm,
    float* __restrict__ C, _Float16* __restrict__ Ch,
    const float* __restrict__ bias, const float* __restrict__ resid,
    int N, int K, int ldch, int act){
  const int wave = threadIdx.x >> 5, lane = threadIdx.x & 31;
  const int tilesN = N >> 4;
  const int tile = blockIdx.x*4 + wave;
  const int tm = tile / tilesN, tn = tile - tm*tilesN;
  const int r  = lane & 15, kg = lane >> 4;
  const _Float16* ap = A  + (long)(tm*16 + r)*K + kg*16;
  const _Float16* bp = Bm + (long)(tn*16 + r)*K + kg*16;
  v8f acc = {};
  for (int k0 = 0; k0 < K; k0 += 32){
    v8h a0 = *(const v8h*)(ap + k0);
    v8h a1 = *(const v8h*)(ap + k0 + 8);
    v8h b0 = *(const v8h*)(bp + k0);
    v8h b1 = *(const v8h*)(bp + k0 + 8);
    v16h av = __builtin_shufflevector(a0, a1, 0,1,2,3,4,5,6,7,8,9,10,11,12,13,14,15);
    v16h bv = __builtin_shufflevector(b0, b1, 0,1,2,3,4,5,6,7,8,9,10,11,12,13,14,15);
    acc = __builtin_amdgcn_wmma_f32_16x16x32_f16(false, av, false, bv,
                                                 (short)0, acc, false, false);
  }
  const int col = tn*16 + r;
  const float bi = bias ? bias[col] : 0.f;
  #pragma unroll
  for (int j = 0; j < 8; j++){
    int row = tm*16 + j + 8*kg;
    float v = acc[j] + bi;
    if (act == 1) v = geluf_(v);
    if (resid)    v += resid[(long)row*N + col];
    C[(long)row*N + col] = v;
    if (Ch) Ch[(long)row*ldch + col] = (_Float16)v;
  }
}

// ---------------- causal depthwise conv (k=4) + SiLU, xz[:, :512] -> u_f16 ----------------

__global__ __launch_bounds__(256) void k_conv_silu(const float* __restrict__ xz,
    const float* __restrict__ cw, _Float16* __restrict__ uh){
  long i = (long)blockIdx.x*256 + threadIdx.x;     // over MTOK*DIM
  int  d = (int)(i % DIM);
  long t = i / DIM;
  int  l = (int)(t % LL);
  long b = t / LL;
  float acc = 0.f;
  #pragma unroll
  for (int k = 0; k < 4; k++){
    int ls = l - 3 + k;
    float u = (ls >= 0) ? xz[((b*LL + ls)*1024) + d] : 0.f;
    acc += u * cw[d*4 + k];
  }
  uh[i] = (_Float16)siluf_(acc);
}

// ---------------- selective scan: fuses dt-proj, softplus, recurrence, gate ----------------

__global__ __launch_bounds__(512) void k_scan(const float* __restrict__ xd,
    const _Float16* __restrict__ uh, const float* __restrict__ xz,
    const float* __restrict__ dtw, const float* __restrict__ dtb,
    const float* __restrict__ alog, const float* __restrict__ Dp,
    _Float16* __restrict__ yh){
  int b = blockIdx.x, d = threadIdx.x;
  float w[RRK], An[NST], s[NST];
  #pragma unroll
  for (int r = 0; r < RRK; r++) w[r] = dtw[d*RRK + r];
  #pragma unroll
  for (int n = 0; n < NST; n++){ An[n] = -__expf(alog[d*NST + n]); s[n] = 0.f; }
  float db = dtb[d], dpv = Dp[d];
  __shared__ float sx[48];                          // dtr[16] | B[16] | C[16]
  for (int l = 0; l < LL; l++){
    long t = (long)b*LL + l;
    if (d < 48) sx[d] = xd[t*48 + d];
    __syncthreads();
    float acc = db;
    #pragma unroll
    for (int r = 0; r < RRK; r++) acc += sx[r]*w[r];
    float delta = softplusf_(acc);
    float u  = (float)uh[t*DIM + d];
    float du = delta*u;
    float y  = 0.f;
    #pragma unroll
    for (int n = 0; n < NST; n++){
      s[n] = __expf(delta*An[n])*s[n] + du*sx[16 + n];
      y   += s[n]*sx[32 + n];
    }
    float z = xz[t*1024 + 512 + d];
    y = (y + dpv*u)*siluf_(z);
    yh[t*DIM + d] = (_Float16)y;
    __syncthreads();
  }
}

// ---------------- final gather of last valid position ----------------

__global__ __launch_bounds__(256) void k_gather(const int* __restrict__ len,
    const float* __restrict__ h, float* __restrict__ out){
  int b = blockIdx.x, d = threadIdx.x;
  int l = len[b] - 1;
  out[b*DD + d] = h[((long)b*LL + l)*DD + d];
}

// ---------------- host ----------------

extern "C" void kernel_launch(void* const* d_in, const int* in_sizes, int n_in,
                              void* d_out, int out_size, void* d_ws, size_t ws_size,
                              hipStream_t stream){
  const int*   seq  = (const int*)  d_in[0];
  const int*   slen = (const int*)  d_in[1];
  const float* emb  = (const float*)d_in[2];
  const float* ln0w = (const float*)d_in[3];
  const float* ln0b = (const float*)d_in[4];
  const float* inw  = (const float*)d_in[5];
  const float* cw   = (const float*)d_in[6];
  const float* xpw  = (const float*)d_in[7];
  const float* dtw  = (const float*)d_in[8];
  const float* dtb  = (const float*)d_in[9];
  const float* alog = (const float*)d_in[10];
  const float* Dpp  = (const float*)d_in[11];
  const float* outw = (const float*)d_in[12];
  const float* mlnw = (const float*)d_in[13];
  const float* mlnb = (const float*)d_in[14];
  const float* f1w  = (const float*)d_in[15];
  const float* f1b  = (const float*)d_in[16];
  const float* f2w  = (const float*)d_in[17];
  const float* f2b  = (const float*)d_in[18];
  float* out = (float*)d_out;

  char* p = (char*)d_ws;
  auto alloc = [&](size_t bytes)->char*{
    char* q = p; p += (bytes + 255) & ~(size_t)255; return q;
  };
  float*    h_f32  = (float*)   alloc((size_t)MTOK*DD*4);
  _Float16* h_f16  = (_Float16*)alloc((size_t)MTOK*DD*2);
  float*    xz     = (float*)   alloc((size_t)MTOK*1024*4);
  _Float16* u_f16  = (_Float16*)alloc((size_t)MTOK*DIM*2);
  float*    xd     = (float*)   alloc((size_t)MTOK*48*4);
  _Float16* y_f16  = (_Float16*)alloc((size_t)MTOK*DIM*2);
  float*    m_f32  = (float*)   alloc((size_t)MTOK*DD*4);
  float*    h2_f32 = (float*)   alloc((size_t)MTOK*DD*4);
  _Float16* h2_f16 = (_Float16*)alloc((size_t)MTOK*DD*2);
  _Float16* f1_f16 = (_Float16*)alloc((size_t)MTOK*32*2);
  float*    f1_f32 = (float*)   alloc((size_t)MTOK*16*4);
  _Float16* w16_in = (_Float16*)alloc((size_t)1024*256*2);
  _Float16* w16_xp = (_Float16*)alloc((size_t)48*512*2);
  _Float16* w16_o  = (_Float16*)alloc((size_t)256*512*2);
  _Float16* w16_f1 = (_Float16*)alloc((size_t)16*256*2);
  _Float16* w16_f2 = (_Float16*)alloc((size_t)256*32*2);

  auto cvt = [&](const float* s, _Float16* d, long n){
    k_f32_to_f16<<<(unsigned)((n + 255)/256), 256, 0, stream>>>(s, d, n);
  };
  auto gemm_small = [&](const _Float16* A, const _Float16* Bw, float* C, _Float16* Ch,
                        const float* bias, const float* resid, int N, int K, int ldch, int act){
    int nt = (MTOK/16) * (N/16);
    k_gemm_f16<<<nt/4, 128, 0, stream>>>(A, Bw, C, Ch, bias, resid, N, K, ldch, act);
  };
  auto gemm_big = [&](const _Float16* A, const _Float16* Bw, float* C, _Float16* Ch,
                      const float* bias, const float* resid, int N, int K, int ldch, int act){
    int blocks = (MTOK/64) * (N/128);
    k_gemm_f16_big<<<blocks, 128, 0, stream>>>(A, Bw, C, Ch, bias, resid, N, K, ldch, act);
  };

  // token embedding + LN0
  k_embed_ln<<<MTOK, 256, 0, stream>>>(seq, emb, ln0w, ln0b, h_f32, h_f16);

  for (int i = 0; i < 2; i++){
    // per-layer weight conversion (f32 -> f16, fc2 K-padded to 32)
    cvt(inw  + (size_t)i*1024*256, w16_in, 1024*256);
    cvt(xpw  + (size_t)i*48*512,   w16_xp, 48*512);
    cvt(outw + (size_t)i*256*512,  w16_o,  256*512);
    cvt(f1w  + (size_t)i*16*256,   w16_f1, 16*256);
    k_pad_fc2<<<(256*32)/256, 256, 0, stream>>>(f2w + (size_t)i*256*16, w16_f2);
    k_fill_f16<<<((long)MTOK*32 + 255)/256, 256, 0, stream>>>(f1_f16, (long)MTOK*32);

    // xz = h @ in_proj_w^T   [8192 x 1024]
    gemm_big(h_f16, w16_in, xz, nullptr, nullptr, nullptr, 1024, 256, 0, 0);
    // causal depthwise conv + SiLU -> u
    k_conv_silu<<<((long)MTOK*DIM)/256, 256, 0, stream>>>(xz, cw + (size_t)i*DIM*4, u_f16);
    // xd = u @ x_proj_w^T    [8192 x 48]
    gemm_small(u_f16, w16_xp, xd, nullptr, nullptr, nullptr, 48, 512, 0, 0);
    // selective scan (fused dt-proj + gate)
    k_scan<<<BB, 512, 0, stream>>>(xd, u_f16, xz,
                                   dtw + (size_t)i*DIM*RRK, dtb + (size_t)i*DIM,
                                   alog + (size_t)i*DIM*NST, Dpp + (size_t)i*DIM, y_f16);
    // m = y @ out_w^T        [8192 x 256]
    gemm_big(y_f16, w16_o, m_f32, nullptr, nullptr, nullptr, 256, 512, 0, 0);
    // h2 = LN(m + h)
    k_ln_resid<<<MTOK, 256, 0, stream>>>(m_f32, h_f32, mlnw + (size_t)i*DD,
                                         mlnb + (size_t)i*DD, h2_f32, h2_f16);
    // f1 = gelu(h2 @ fc1^T + b1)   [8192 x 16], f16 mirror padded to ld 32
    gemm_small(h2_f16, w16_f1, f1_f32, f1_f16, f1b + (size_t)i*16, nullptr, 16, 256, 32, 1);
    // h = f1 @ fc2^T + b2 + h2     [8192 x 256], K padded 16->32
    gemm_big(f1_f16, w16_f2, h_f32, h_f16, f2b + (size_t)i*DD, h2_f32, 256, 32, 256, 0);
  }

  k_gather<<<BB, 256, 0, stream>>>(slen, h_f32, out);
  (void)in_sizes; (void)n_in; (void)out_size; (void)ws_size;
}
